// Involution_54004918780590
// MI455X (gfx1250) — matrix operand order repaired and checked by reading, not compiled, not run on hardware
//
#include <hip/hip_runtime.h>
#include <hip/hip_bf16.h>
#include <math.h>

#define CC   256
#define HH   64
#define WW   64
#define BB   4
#define HID  64
#define TAPS 49
#define PADK 3
#define NROW (TAPS * CC)          // 12544 kernel-projection rows
#define LROW 72                   // padded LDS row stride (bf16) -> conflict-free b128
#define LPLANE (TAPS * LROW)      // 3528 bf16 per plane

typedef __attribute__((ext_vector_type(2)))  float  v2f;
typedef __attribute__((ext_vector_type(8)))  float  v8f;
typedef __attribute__((ext_vector_type(16))) __bf16 v16bf;

union FragBF { uint4 u[2]; v16bf v; };

__device__ __forceinline__ v8f wmma_f32(v2f a, v2f b, v8f c) {
  return __builtin_amdgcn_wmma_f32_16x16x4_f32(false, a, false, b, (short)0, c,
                                               false, false);
}
__device__ __forceinline__ v8f wmma_bf16(v16bf a, v16bf b, v8f c) {
  return __builtin_amdgcn_wmma_f32_16x16x32_bf16(false, a, false, b, (short)0, c,
                                                 false, false);
}

// ---------------------------------------------------------------------------
// Split kproj_w into bf16 hi/lo planes, regrouped per involution group:
// kws layout = [g][plane(hi,lo)][49][64] bf16  (contiguous 12.25 KB per group)
// ---------------------------------------------------------------------------
__global__ __launch_bounds__(256) void kw_split_kernel(
    const float* __restrict__ kw, __bf16* __restrict__ kws)
{
  const int e = blockIdx.x * 256 + threadIdx.x;     // over NROW*HID
  if (e >= NROW * HID) return;
  const int row = e >> 6, k = e & 63;
  const int g = row / TAPS;
  const int r = row - g * TAPS;
  const float v  = kw[e];
  const __bf16 h = (__bf16)v;
  const __bf16 l = (__bf16)(v - (float)h);
  kws[(((size_t)g * 2 + 0) * TAPS + r) * HID + k] = h;
  kws[(((size_t)g * 2 + 1) * TAPS + r) * HID + k] = l;
}

// ---------------------------------------------------------------------------
// Stage A: h = ReLU(BN(reduce_w @ x)), fp32 WMMA (tiny: 0.54 GFLOP).
// Output: bf16 hi/lo planes, pixel-major (B*H*W, HID) for contiguous B-frags.
// ---------------------------------------------------------------------------
__global__ __launch_bounds__(256) void involution_stage_a(
    const float* __restrict__ x, const float* __restrict__ rw,
    const float* __restrict__ gamma, const float* __restrict__ beta,
    const float* __restrict__ mean, const float* __restrict__ var,
    __bf16* __restrict__ hhi, __bf16* __restrict__ hlo)
{
  const int wave = threadIdx.x >> 5;
  const int lane = threadIdx.x & 31;
  const int n    = lane & 15;
  const int hi   = lane >> 4;

  const int tile = blockIdx.x * 8 + wave;     // 0..1023
  const int b    = tile >> 8;
  const int hh   = (tile & 255) >> 2;
  const int w0   = (tile & 3) << 4;
  const int pixw = w0 + n;

  v8f acc[4] = {};
  const float* xb = x + ((size_t)b * CC * HH * WW) + (size_t)hh * WW + pixw;
  for (int kk = 0; kk < 64; ++kk) {
    const int c0 = kk * 4 + hi * 2;
    v2f bf;
    bf.x = xb[(size_t)c0 * (HH * WW)];
    bf.y = xb[(size_t)(c0 + 1) * (HH * WW)];
#pragma unroll
    for (int T = 0; T < 4; ++T) {
      const float* ap = rw + (size_t)(T * 16 + n) * CC + c0;
      v2f af; af.x = ap[0]; af.y = ap[1];
      acc[T] = wmma_f32(af, bf, acc[T]);
    }
  }

  const size_t pix = ((size_t)b * HH + hh) * WW + pixw;
#pragma unroll
  for (int T = 0; T < 4; ++T) {
#pragma unroll
    for (int r = 0; r < 8; ++r) {
      const int o = T * 16 + r + hi * 8;
      const float inv = gamma[o] * rsqrtf(var[o] + 1e-5f);
      float v = acc[T][r] * inv + (beta[o] - mean[o] * inv);
      v = v > 0.f ? v : 0.f;
      const __bf16 hb = (__bf16)v;
      hhi[pix * HID + o] = hb;
      hlo[pix * HID + o] = (__bf16)(v - (float)hb);
    }
  }
}

// ---------------------------------------------------------------------------
// Stage B (fused): ker = kproj_w @ h + b via bf16x3 WMMA; zero-mean + L2 norm
// over 49 taps; 7x7 depthwise involution. Block = 8 waves x one 16-pixel tile
// each; the per-group 49x64 hi/lo kproj tile is pushed into LDS once per block
// with GLOBAL_LOAD_ASYNC_TO_LDS_B128 (ASYNCcnt) and shared by all 8 waves.
// ---------------------------------------------------------------------------
__global__ __launch_bounds__(256) void involution_stage_b(
    const float* __restrict__ x, const float* __restrict__ kb,
    const __bf16* __restrict__ hhi, const __bf16* __restrict__ hlo,
    const __bf16* __restrict__ kws, float* __restrict__ out)
{
  __shared__ __align__(16) __bf16 lA[2 * LPLANE];   // 14.1 KB: hi plane, lo plane

  const int wave = threadIdx.x >> 5;
  const int lane = threadIdx.x & 31;
  const int n    = lane & 15;
  const int hi   = lane >> 4;

  const int tile = blockIdx.x * 8 + wave;     // 0..1023
  const int b    = tile >> 8;
  const int hh   = (tile & 255) >> 2;
  const int w0   = (tile & 3) << 4;
  const int pixw = w0 + n;
  const size_t pix = ((size_t)b * HH + hh) * WW + pixw;

  // B fragments (reused across all groups): per lane 16 consecutive K
  // starting at chunk*32 + hi*16 (contiguous k order, same as A: lanes 0-15 K=0..15, 16-31 K=16..31).
  FragBF Bh[2], Bl[2];
#pragma unroll
  for (int ch = 0; ch < 2; ++ch) {
    const uint4* ph = (const uint4*)(hhi + pix * HID + ch * 32 + hi * 16);
    Bh[ch].u[0] = ph[0]; Bh[ch].u[1] = ph[1];
    const uint4* pl = (const uint4*)(hlo + pix * HID + ch * 32 + hi * 16);
    Bl[ch].u[0] = pl[0]; Bl[ch].u[1] = pl[1];
  }

  const float* xg0 = x + (size_t)b * CC * HH * WW;
  const uint32_t lbase = (uint32_t)(uintptr_t)lA;   // low 32 bits = LDS offset
  const int g0 = blockIdx.y * 64;

  for (int g = g0; g < g0 + 64; ++g) {
    __syncthreads();   // previous iteration's LDS reads complete
    // Async fill: 784 x 16B direct global->LDS (hi plane then lo plane).
    {
      const uint64_t gbase =
          (uint64_t)(uintptr_t)kws + (uint64_t)g * (2u * TAPS * HID * 2u);
      for (int idx = threadIdx.x; idx < 2 * TAPS * 8; idx += 256) {
        const int p   = idx / (TAPS * 8);
        const int rem = idx - p * (TAPS * 8);
        const int row = rem >> 3, ch = rem & 7;
        const uint32_t ldst = lbase +
            (uint32_t)(p * LPLANE + row * LROW + ch * 8) * 2u;
        const uint32_t goff =
            (uint32_t)((p * TAPS + row) * HID + ch * 8) * 2u;
        asm volatile("global_load_async_to_lds_b128 %0, %1, %2"
                     :: "v"(ldst), "v"(goff), "s"(gbase) : "memory");
      }
      asm volatile("s_wait_asynccnt 0" ::: "memory");
    }
    __syncthreads();

    // GEMM: 4 row-tiles x 2 K-chunks x 3 bf16 terms = 24 WMMA.
    v8f acc[4] = {};
#pragma unroll
    for (int T = 0; T < 4; ++T) {
      int row = T * 16 + n;
      if (row > TAPS - 1) row = TAPS - 1;     // rows 49..63 unused, clamp
      const __bf16* la = lA + row * LROW;
#pragma unroll
      for (int ch = 0; ch < 2; ++ch) {
        // A layout per lane: K in {K0..K0+15}, K0 = ch*32+hi*16
        FragBF Ah, Al;
        const uint4* p0 = (const uint4*)(la + ch * 32 + hi * 16);
        Ah.u[0] = p0[0]; Ah.u[1] = p0[1];
        const uint4* p1 = (const uint4*)(la + LPLANE + ch * 32 + hi * 16);
        Al.u[0] = p1[0]; Al.u[1] = p1[1];
        acc[T] = wmma_bf16(Ah.v, Bh[ch].v, acc[T]);
        acc[T] = wmma_bf16(Ah.v, Bl[ch].v, acc[T]);
        acc[T] = wmma_bf16(Al.v, Bh[ch].v, acc[T]);
      }
    }

    // Bias + tap mean over the 49 valid taps. Slot taps are the compile-time
    // pair {t0, t0+8} selected by lane half -> single cndmask, no div chains.
    const float* kbg = kb + (size_t)g * TAPS;
    float s = 0.f;
#pragma unroll
    for (int T = 0; T < 4; ++T) {
#pragma unroll
      for (int r = 0; r < 8; ++r) {
        const int t0 = T * 16 + r;
        const int t1 = t0 + 8;
        float v = 0.f;
        if (t0 < TAPS) {
          const bool v1 = (t1 < TAPS);              // compile-time
          const int  t  = hi ? (v1 ? t1 : 0) : t0;  // cndmask of constants
          const bool lv = (hi == 0) | v1;
          if (lv) v = acc[T][r] + kbg[t];
        }
        acc[T][r] = v;
        s += v;
      }
    }
    s += __shfl_xor(s, 16, 32);
    const float mu = s * (1.f / 49.f);

    float ss = 0.f;
#pragma unroll
    for (int T = 0; T < 4; ++T) {
#pragma unroll
      for (int r = 0; r < 8; ++r) {
        const int t0 = T * 16 + r;
        if (t0 < TAPS) {
          const bool lv = (hi == 0) | (t0 + 8 < TAPS);
          if (lv) { const float cv = acc[T][r] - mu; ss += cv * cv; }
        }
      }
    }
    ss += __shfl_xor(ss, 16, 32);
    float dn = sqrtf(ss);
    dn = dn > 1e-6f ? dn : 1e-6f;
    const float rn = 1.f / dn;

    // Involution taps against x (fully L2-resident).
    const float* xgc = xg0 + (size_t)g * HH * WW;
    float oacc = 0.f;
#pragma unroll
    for (int T = 0; T < 4; ++T) {
#pragma unroll
      for (int r = 0; r < 8; ++r) {
        const int t0 = T * 16 + r;
        const int t1 = t0 + 8;
        if (t0 < TAPS) {
          const bool v1 = (t1 < TAPS);                      // compile-time
          const int  i  = hi ? (v1 ? t1 / 7 : 0) : t0 / 7;  // cndmask consts
          const int  j  = hi ? (v1 ? t1 % 7 : 0) : t0 % 7;
          const bool lv = (hi == 0) | v1;
          const int sh = hh + i - PADK;
          const int sw = pixw + j - PADK;
          float xv = 0.f;
          if (lv && (unsigned)sh < (unsigned)HH && (unsigned)sw < (unsigned)WW)
            xv = xgc[sh * WW + sw];
          oacc += (acc[T][r] - mu) * xv;
        }
      }
    }
    oacc += __shfl_xor(oacc, 16, 32);
    if (hi == 0)
      out[(((size_t)b * CC + g) * HH + hh) * WW + pixw] = oacc * rn;
  }
}

extern "C" void kernel_launch(void* const* d_in, const int* in_sizes, int n_in,
                              void* d_out, int out_size, void* d_ws, size_t ws_size,
                              hipStream_t stream) {
  const float* x  = (const float*)d_in[0];
  const float* rw = (const float*)d_in[1];
  const float* gm = (const float*)d_in[2];
  const float* bt = (const float*)d_in[3];
  const float* mn = (const float*)d_in[4];
  const float* vr = (const float*)d_in[5];
  const float* kw = (const float*)d_in[6];
  const float* kb = (const float*)d_in[7];
  float* out = (float*)d_out;

  // Workspace layout (bf16): h_hi | h_lo | kproj hi/lo split  (~7.1 MiB)
  const size_t NPIX = (size_t)BB * HH * WW;
  __bf16* hhi = (__bf16*)d_ws;
  __bf16* hlo = hhi + NPIX * HID;
  __bf16* kws = hlo + NPIX * HID;

  kw_split_kernel<<<dim3((NROW * HID + 255) / 256), dim3(256), 0, stream>>>(kw, kws);
  involution_stage_a<<<dim3(128), dim3(256), 0, stream>>>(x, rw, gm, bt, mn, vr,
                                                          hhi, hlo);
  involution_stage_b<<<dim3(128, 4), dim3(256), 0, stream>>>(x, kb, hhi, hlo,
                                                             kws, out);
}
